// TrainTest_55662776156779
// MI455X (gfx1250) — matrix operand order repaired and checked by reading, not compiled, not run on hardware
//
#include <hip/hip_runtime.h>
#include <hip/hip_bf16.h>
#include <math.h>

typedef __attribute__((ext_vector_type(16))) __bf16 v16bf;
typedef __attribute__((ext_vector_type(8)))  __bf16 v8bf;
typedef __attribute__((ext_vector_type(8)))  float  v8f;

#define E_DIM   1024
#define FOUR_E  4096
#define B_N     8
#define T_FULL  512
#define T_M1    511
#define ROWS    4088          // B_N * T_M1
#define VOCAB_N 1024

// ---------------- fragment loaders (WMMA 16x16x32 bf16; this code's contiguous k order, same in A and B) -------
// A-matrix 16x32 bf16: lane m=lane&15, hi=lane>>4;
//   elems 0..7  = K = k0 + hi*16     .. +8   (contiguous)
//   elems 8..15 = K = k0 + hi*16 + 8 .. +8   (contiguous)
__device__ __forceinline__ v16bf load_a_frag(const __bf16* row, int k0, int hi) {
    union { v16bf v; v8bf h[2]; } u;
    u.h[0] = *reinterpret_cast<const v8bf*>(row + k0 + hi * 16);
    u.h[1] = *reinterpret_cast<const v8bf*>(row + k0 + hi * 16 + 8);
    return u.v;
}
// B-matrix 32x16 bf16: lane n=lane&15, hi=lane>>4; elems i = K = k0+hi*16+i.
// wtrow points at transposed-weight row n (K contiguous).
__device__ __forceinline__ v16bf load_b_frag(const __bf16* wtrow, int k0, int hi) {
    union { v16bf v; v8bf h[2]; } u;
    const v8bf* p = reinterpret_cast<const v8bf*>(wtrow + k0 + hi * 16);
    u.h[0] = p[0];
    u.h[1] = p[1];
    return u.v;
}

__device__ __forceinline__ v8f wmma_bf16(v16bf a, v16bf b, v8f c) {
    return __builtin_amdgcn_wmma_f32_16x16x32_bf16(false, a, false, b, (short)0, c, false, false);
}

__device__ __forceinline__ float sigm(float x) { return 1.0f / (1.0f + __expf(-x)); }

// ---------------- prep kernels ----------------------------------------------
__global__ void cvt_bf16_kernel(const float* __restrict__ src, __bf16* __restrict__ dst, int n) {
    int i = blockIdx.x * blockDim.x + threadIdx.x;
    if (i < n) dst[i] = (__bf16)src[i];
}

// dst[c*rows + r] = src[r*cols + c]  (fp32 [rows][cols] -> bf16 [cols][rows])
__global__ void transpose_bf16_kernel(const float* __restrict__ src, __bf16* __restrict__ dst,
                                      int rows, int cols) {
    int i = blockIdx.x * blockDim.x + threadIdx.x;
    if (i < rows * cols) {
        int r = i / cols;
        int c = i - r * cols;
        dst[(size_t)c * rows + r] = (__bf16)src[i];
    }
}

__global__ void init_state_kernel(const float* __restrict__ c0, const float* __restrict__ h0,
                                  __bf16* __restrict__ hb0, __bf16* __restrict__ hb1,
                                  float* __restrict__ c_state, float* __restrict__ accum) {
    int i = blockIdx.x * blockDim.x + threadIdx.x;   // 16*1024 threads
    if (i < 16 * E_DIM) {
        int m = i >> 10;
        int e = i & (E_DIM - 1);
        __bf16 hv = (m < B_N) ? (__bf16)h0[e] : (__bf16)0.0f;  // rows 8..15 zero pad
        hb0[i] = hv;
        hb1[i] = hv;
        if (m < B_N) c_state[m * E_DIM + e] = c0[e];
    }
    if (i < 2) accum[i] = 0.0f;
}

// ---------------- generic 16x64-per-wave GEMM body --------------------------
// Each wave: 16 rows x 64 cols, K = E_DIM. A fragment loaded once per k-step,
// reused across 4 WMMAs. Double-buffered (manual unroll-by-2, fragments
// reloaded in place): no register-rotation copies, loads for one buffer fly
// while the other buffer's WMMAs execute.
struct Acc4 { v8f a0, a1, a2, a3; };

__device__ __forceinline__ Acc4 gemm_16x64(const __bf16* __restrict__ arow,
                                           const __bf16* __restrict__ brow0, int hi) {
    Acc4 c = {{}, {}, {}, {}};
    const __bf16* br1 = brow0 + 16 * E_DIM;
    const __bf16* br2 = brow0 + 32 * E_DIM;
    const __bf16* br3 = brow0 + 48 * E_DIM;

    // set0 <- k=0, set1 <- k=32
    v16bf a0 = load_a_frag(arow, 0, hi);
    v16bf p0 = load_b_frag(brow0, 0, hi);
    v16bf p1 = load_b_frag(br1, 0, hi);
    v16bf p2 = load_b_frag(br2, 0, hi);
    v16bf p3 = load_b_frag(br3, 0, hi);
    v16bf a1 = load_a_frag(arow, 32, hi);
    v16bf q0 = load_b_frag(brow0, 32, hi);
    v16bf q1 = load_b_frag(br1, 32, hi);
    v16bf q2 = load_b_frag(br2, 32, hi);
    v16bf q3 = load_b_frag(br3, 32, hi);

#pragma unroll 1
    for (int k0 = 64; k0 < E_DIM; k0 += 64) {
        c.a0 = wmma_bf16(a0, p0, c.a0);
        c.a1 = wmma_bf16(a0, p1, c.a1);
        c.a2 = wmma_bf16(a0, p2, c.a2);
        c.a3 = wmma_bf16(a0, p3, c.a3);
        a0 = load_a_frag(arow, k0, hi);
        p0 = load_b_frag(brow0, k0, hi);
        p1 = load_b_frag(br1, k0, hi);
        p2 = load_b_frag(br2, k0, hi);
        p3 = load_b_frag(br3, k0, hi);
        c.a0 = wmma_bf16(a1, q0, c.a0);
        c.a1 = wmma_bf16(a1, q1, c.a1);
        c.a2 = wmma_bf16(a1, q2, c.a2);
        c.a3 = wmma_bf16(a1, q3, c.a3);
        a1 = load_a_frag(arow, k0 + 32, hi);
        q0 = load_b_frag(brow0, k0 + 32, hi);
        q1 = load_b_frag(br1, k0 + 32, hi);
        q2 = load_b_frag(br2, k0 + 32, hi);
        q3 = load_b_frag(br3, k0 + 32, hi);
    }
    c.a0 = wmma_bf16(a0, p0, c.a0);
    c.a1 = wmma_bf16(a0, p1, c.a1);
    c.a2 = wmma_bf16(a0, p2, c.a2);
    c.a3 = wmma_bf16(a0, p3, c.a3);
    c.a0 = wmma_bf16(a1, q0, c.a0);
    c.a1 = wmma_bf16(a1, q1, c.a1);
    c.a2 = wmma_bf16(a1, q2, c.a2);
    c.a3 = wmma_bf16(a1, q3, c.a3);
    return c;
}

// ---------------- xproj = gather(X, ids) @ W1 + bias   (parallel, WMMA) -----
// grid: x = 8 (8 waves x 64-col strips), y = 256 row-tiles; block = 256
__global__ void xproj_kernel(const int* __restrict__ data, const __bf16* __restrict__ Xb,
                             const __bf16* __restrict__ W1T, const float* __restrict__ bias,
                             float* __restrict__ xproj) {
    int wv = threadIdx.x >> 5;
    int lane = threadIdx.x & 31;
    int hi = lane >> 4;
    int ln = lane & 15;
    int n0 = (blockIdx.x * 8 + wv) * 64;
    int r0 = blockIdx.y * 16;

    int r = r0 + ln;
    if (r > ROWS - 1) r = ROWS - 1;
    int t = r >> 3, bb = r & 7;
    int id = data[bb * T_FULL + t];                       // x_ids[b,t]
    const __bf16* arow = Xb + (size_t)id * E_DIM;
    const __bf16* brow = W1T + (size_t)(n0 + ln) * E_DIM;

    Acc4 c = gemm_16x64(arow, brow, hi);
    v8f acc[4] = {c.a0, c.a1, c.a2, c.a3};

    bool full = (r0 + 16) <= ROWS;
#pragma unroll
    for (int g = 0; g < 4; g++) {
        int col = n0 + g * 16 + ln;
        float bv = bias[col];
        float* outc = xproj + col;
        if (full) {
#pragma unroll
            for (int rr = 0; rr < 8; rr++)
                outc[(size_t)(r0 + rr + hi * 8) * FOUR_E] = acc[g][rr] + bv;
        } else {
#pragma unroll
            for (int rr = 0; rr < 8; rr++) {
                int row = r0 + rr + hi * 8;               // C layout: M = rr + hi*8, N = ln
                if (row < ROWS) outc[(size_t)row * FOUR_E] = acc[g][rr] + bv;
            }
        }
    }
}

// ---------------- one LSTM timestep (serial chain, launched 511x) -----------
// grid = 64 blocks (16 cells each), block = 128 (4 waves, one per gate i/j/f/o)
__global__ void lstm_step_kernel(const __bf16* __restrict__ hin, __bf16* __restrict__ hout,
                                 float* __restrict__ c_state, const float* __restrict__ xproj,
                                 const __bf16* __restrict__ W2T, __bf16* __restrict__ hs,
                                 int t) {
    __shared__ float res[4][B_N][16];
    int wv = threadIdx.x >> 5;        // gate index 0..3  (i, j, f, o)
    int lane = threadIdx.x & 31;
    int hi = lane >> 4;
    int ln = lane & 15;
    int cb = blockIdx.x * 16;         // cell base
    int n0 = wv * E_DIM + cb;         // columns of this gate

    const __bf16* arow = hin + (size_t)ln * E_DIM;            // rows 8..15 are zeros
    const __bf16* brow = W2T + (size_t)(n0 + ln) * E_DIM;

    v8f acc = {};
    v16bf a0 = load_a_frag(arow, 0, hi);
    v16bf b0 = load_b_frag(brow, 0, hi);
    v16bf a1 = load_a_frag(arow, 32, hi);
    v16bf b1 = load_b_frag(brow, 32, hi);
#pragma unroll 1
    for (int k0 = 64; k0 < E_DIM; k0 += 64) {
        acc = wmma_bf16(a0, b0, acc);
        a0 = load_a_frag(arow, k0, hi);
        b0 = load_b_frag(brow, k0, hi);
        acc = wmma_bf16(a1, b1, acc);
        a1 = load_a_frag(arow, k0 + 32, hi);
        b1 = load_b_frag(brow, k0 + 32, hi);
    }
    acc = wmma_bf16(a0, b0, acc);
    acc = wmma_bf16(a1, b1, acc);

    if (hi == 0) {                    // batch rows 0..7 live in lanes 0..15
        int col = n0 + ln;
#pragma unroll
        for (int rr = 0; rr < B_N; rr++)
            res[wv][rr][ln] = acc[rr] + xproj[(size_t)(t * B_N + rr) * FOUR_E + col];
    }
    __syncthreads();

    // 128 threads == 8 batches x 16 cells
    int m = threadIdx.x >> 4;
    int e = threadIdx.x & 15;
    float iv = res[0][m][e];
    float jv = res[1][m][e];
    float fv = res[2][m][e];
    float ov = res[3][m][e];
    int ci = m * E_DIM + cb + e;
    float cn = c_state[ci] * sigm(fv) + tanhf(jv) * sigm(iv);
    float hn = tanhf(cn) * sigm(ov);
    c_state[ci] = cn;
    hout[ci] = (__bf16)hn;
    hs[(size_t)(t * B_N + m) * E_DIM + cb + e] = (__bf16)hn;
}

// ---------------- logits = hs @ Vw  (parallel, WMMA) ------------------------
// grid: x = 2 (8 waves x 64-col strips), y = 256 row-tiles; block = 256
__global__ void logits_kernel(const __bf16* __restrict__ hs, const __bf16* __restrict__ VwT,
                              float* __restrict__ logits) {
    int wv = threadIdx.x >> 5;
    int lane = threadIdx.x & 31;
    int hi = lane >> 4;
    int ln = lane & 15;
    int n0 = (blockIdx.x * 8 + wv) * 64;
    int r0 = blockIdx.y * 16;

    int r = r0 + ln;
    if (r > ROWS - 1) r = ROWS - 1;
    const __bf16* arow = hs + (size_t)r * E_DIM;
    const __bf16* brow = VwT + (size_t)(n0 + ln) * E_DIM;

    Acc4 c = gemm_16x64(arow, brow, hi);
    v8f acc[4] = {c.a0, c.a1, c.a2, c.a3};

    bool full = (r0 + 16) <= ROWS;
#pragma unroll
    for (int g = 0; g < 4; g++) {
        int col = n0 + g * 16 + ln;
        float* outc = logits + col;
        if (full) {
#pragma unroll
            for (int rr = 0; rr < 8; rr++)
                outc[(size_t)(r0 + rr + hi * 8) * VOCAB_N] = acc[g][rr];
        } else {
#pragma unroll
            for (int rr = 0; rr < 8; rr++) {
                int row = r0 + rr + hi * 8;
                if (row < ROWS) outc[(size_t)row * VOCAB_N] = acc[g][rr];
            }
        }
    }
}

// ---------------- per-row loss + top-k -------------------------------------
// grid = 4088 blocks, block = 256
__global__ void loss_kernel(const float* __restrict__ logits, const int* __restrict__ data,
                            const int* __restrict__ topk_p, float* __restrict__ accum) {
    __shared__ float sf[256];
    __shared__ int si[256];
    int r = blockIdx.x;
    int t = r >> 3, bb = r & 7;
    int label = data[bb * T_FULL + t + 1];
    const float* row = logits + (size_t)r * VOCAB_N;
    float gold = row[label];

    float mx = -1e30f;
    for (int i = threadIdx.x; i < VOCAB_N; i += 256) mx = fmaxf(mx, row[i]);
    sf[threadIdx.x] = mx;
    __syncthreads();
    for (int s = 128; s > 0; s >>= 1) {
        if (threadIdx.x < s) sf[threadIdx.x] = fmaxf(sf[threadIdx.x], sf[threadIdx.x + s]);
        __syncthreads();
    }
    mx = sf[0];
    __syncthreads();

    float se = 0.0f;
    int gt = 0;
    for (int i = threadIdx.x; i < VOCAB_N; i += 256) {
        float v = row[i];
        se += __expf(v - mx);
        gt += (v > gold) ? 1 : 0;
    }
    sf[threadIdx.x] = se;
    si[threadIdx.x] = gt;
    __syncthreads();
    for (int s = 128; s > 0; s >>= 1) {
        if (threadIdx.x < s) {
            sf[threadIdx.x] += sf[threadIdx.x + s];
            si[threadIdx.x] += si[threadIdx.x + s];
        }
        __syncthreads();
    }
    if (threadIdx.x == 0) {
        atomicAdd(&accum[0], (mx + logf(sf[0])) - gold);
        if (si[0] < *topk_p) atomicAdd(&accum[1], 1.0f);
    }
}

__global__ void finalize_kernel(const float* __restrict__ accum, float* __restrict__ out) {
    out[0] = accum[0];
    out[1] = accum[1] * (1.0f / (float)ROWS);
}

// ---------------- host launcher --------------------------------------------
extern "C" void kernel_launch(void* const* d_in, const int* in_sizes, int n_in,
                              void* d_out, int out_size, void* d_ws, size_t ws_size,
                              hipStream_t stream) {
    const int*   data = (const int*)d_in[0];
    const int*   topk = (const int*)d_in[1];
    const float* X    = (const float*)d_in[2];
    const float* w    = (const float*)d_in[3];
    const float* bias = (const float*)d_in[4];
    const float* Vw   = (const float*)d_in[5];
    const float* c0   = (const float*)d_in[6];
    const float* h0   = (const float*)d_in[7];
    float* out = (float*)d_out;

    // workspace carve (256B aligned)
    char* p = (char*)d_ws;
    auto carve = [&](size_t bytes) -> void* {
        void* q = (void*)p;
        p += (bytes + 255) & ~(size_t)255;
        return q;
    };
    __bf16* Xb    = (__bf16*)carve((size_t)VOCAB_N * E_DIM * 2);
    __bf16* W1T   = (__bf16*)carve((size_t)FOUR_E * E_DIM * 2);
    __bf16* W2T   = (__bf16*)carve((size_t)FOUR_E * E_DIM * 2);
    __bf16* VwT   = (__bf16*)carve((size_t)VOCAB_N * E_DIM * 2);
    float*  xproj = (float*) carve((size_t)ROWS * FOUR_E * 4);
    __bf16* hs    = (__bf16*)carve((size_t)ROWS * E_DIM * 2);
    float*  lgts  = (float*) carve((size_t)ROWS * VOCAB_N * 4);
    __bf16* hb0   = (__bf16*)carve((size_t)16 * E_DIM * 2);
    __bf16* hb1   = (__bf16*)carve((size_t)16 * E_DIM * 2);
    float*  cst   = (float*) carve((size_t)B_N * E_DIM * 4);
    float*  accum = (float*) carve(256);

    // 1) precision conversion + transposes (bf16, K-contiguous B operands)
    {
        int n = VOCAB_N * E_DIM;
        cvt_bf16_kernel<<<(n + 255) / 256, 256, 0, stream>>>(X, Xb, n);
    }
    transpose_bf16_kernel<<<(E_DIM * FOUR_E + 255) / 256, 256, 0, stream>>>(
        w, W1T, E_DIM, FOUR_E);                       // rows 0..E-1 of w
    transpose_bf16_kernel<<<(E_DIM * FOUR_E + 255) / 256, 256, 0, stream>>>(
        w + (size_t)E_DIM * FOUR_E, W2T, E_DIM, FOUR_E);
    transpose_bf16_kernel<<<(E_DIM * VOCAB_N + 255) / 256, 256, 0, stream>>>(
        Vw, VwT, E_DIM, VOCAB_N);
    init_state_kernel<<<64, 256, 0, stream>>>(c0, h0, hb0, hb1, cst, accum);

    // 2) hoisted input projection: (4088 x 1024) @ (1024 x 4096) + b, WMMA bf16
    xproj_kernel<<<dim3(FOUR_E / 64 / 8, (ROWS + 15) / 16), 256, 0, stream>>>(
        data, Xb, W1T, bias, xproj);

    // 3) serial recurrence: 511 small WMMA kernels, w2 slice stays in L2
    for (int t = 0; t < T_M1; ++t) {
        const __bf16* hin = (t & 1) ? hb1 : hb0;
        __bf16* hout      = (t & 1) ? hb0 : hb1;
        lstm_step_kernel<<<E_DIM / 16, 128, 0, stream>>>(hin, hout, cst, xproj, W2T, hs, t);
    }

    // 4) logits GEMM (4088 x 1024) @ (1024 x 1024), WMMA bf16
    logits_kernel<<<dim3(VOCAB_N / 64 / 8, (ROWS + 15) / 16), 256, 0, stream>>>(hs, VwT, lgts);

    // 5) loss + top-k, 6) finalize
    loss_kernel<<<ROWS, 256, 0, stream>>>(lgts, data, topk, accum);
    finalize_kernel<<<1, 1, 0, stream>>>(accum, out);
}
